// SelfAttention_15736760172863
// MI455X (gfx1250) — compile-verified, matrix operand order checked by reading, not run on hardware
//
#include <hip/hip_runtime.h>
#include <stdint.h>

// ---------------------------------------------------------------------------
// CDNA5 (gfx1250) wave32 WMMA self-attention forward.
// B=2, H=16, S=2048, D=64, E=1024. bf16 WMMA, f32 accumulate.
// Projection GEMMs: 32x64 tiles/wave, weight tiles staged block-wide in LDS
// via global_load_async_to_lds_b128 (ASYNCcnt double buffering). Attention:
// flash-style recompute, softmax stats via __shfl_xor, P-tile re-fragmented
// through LDS. __launch_bounds__(128,1): full VGPR budget, no spills.
// ---------------------------------------------------------------------------

typedef __attribute__((ext_vector_type(16))) __bf16 v16bf;
typedef __attribute__((ext_vector_type(8)))  float  v8f;
typedef __attribute__((ext_vector_type(4)))  int    v4i;

union Frag16 { v16bf v; uint32_t u[8]; uint4 q[2]; };

__device__ __forceinline__ v8f zero8() {
  v8f z = {0.f, 0.f, 0.f, 0.f, 0.f, 0.f, 0.f, 0.f};
  return z;
}

__device__ __forceinline__ v8f wmma_bf16(v16bf a, v16bf b, v8f c) {
  return __builtin_amdgcn_wmma_f32_16x16x32_bf16(false, a, false, b,
                                                 (short)0, c, false, false);
}

// 16x32 bf16 A-fragment (and B as A-layout of B^T): two contiguous 16B runs
// per lane (K = half*8..+7 and 16+half*8..+7). Needs ld%8==0, 16B alignment.
__device__ __forceinline__ v16bf load_frag16(const uint16_t* __restrict__ base,
                                             int ld, int lane) {
  const int r    = lane & 15;
  const int half = lane >> 4;
  const uint16_t* p = base + r * ld + half * 8;
  Frag16 f;
  f.q[0] = *(const uint4*)(p);
  f.q[1] = *(const uint4*)(p + 16);
  return f.v;
}

__device__ __forceinline__ uint16_t f2bf(float x) {  // RNE f32 -> bf16
  uint32_t u = __float_as_uint(x);
  u += 0x7FFFu + ((u >> 16) & 1u);
  return (uint16_t)(u >> 16);
}

// ---- CDNA5 async global->LDS copy (16 bytes / lane) -----------------------
typedef __attribute__((address_space(1))) v4i* g4p;   // global int4*
typedef __attribute__((address_space(3))) v4i* l4p;   // LDS int4*

__device__ __forceinline__ void async_copy16(const uint16_t* g, uint16_t* l) {
#if __has_builtin(__builtin_amdgcn_global_load_async_to_lds_b128)
  __builtin_amdgcn_global_load_async_to_lds_b128((g4p)g, (l4p)l, 0, 0);
#else
  *(uint4*)l = *(const uint4*)g;   // sync fallback
#endif
}
__device__ __forceinline__ void wait_async_le2() {
  asm volatile("s_wait_asynccnt 2" ::: "memory");
}
__device__ __forceinline__ void wait_async_0() {
  asm volatile("s_wait_asynccnt 0" ::: "memory");
}

// Stage W[n0..n0+63][k..k+31] (ld=1024) into lbuf[64][32]; 128 threads,
// 2 x b128 per thread.
__device__ __forceinline__ void stage_w(const uint16_t* __restrict__ W,
                                        uint16_t* lbuf, int n0, int k,
                                        int tid) {
#pragma unroll
  for (int c = tid; c < 256; c += 128) {
    const int row = c >> 2;
    const int col = (c & 3) * 8;
    async_copy16(W + (size_t)(n0 + row) * 1024 + k + col,
                 lbuf + row * 32 + col);
  }
}

// ---------------------------------------------------------------------------
__global__ void k_f32_to_bf16(const float* __restrict__ src,
                              uint16_t* __restrict__ dst, int n) {
  int i = blockIdx.x * blockDim.x + threadIdx.x;
  if (i < n) dst[i] = f2bf(src[i]);
}

// ---------------------------------------------------------------------------
// QKV projection: out[m,n] = sum_k X[m,k]*W[n,k].
// grid (4096/128, 1024/64, 3), block 128. Wave = 32x64 tile. Weight tile in
// LDS (async, double buffered); A prefetch issued after the barrier so its
// latency overlaps the WMMA block; B fragments batch-loaded from LDS.
__global__ void __launch_bounds__(128, 1)
k_qkv(const uint16_t* __restrict__ Xb,   // [4096,1024]
      const uint16_t* __restrict__ Wqb,
      const uint16_t* __restrict__ Wkb,
      const uint16_t* __restrict__ Wvb,
      uint16_t* __restrict__ Qb,         // [B,H,S,D]
      uint16_t* __restrict__ Kb,         // [B,H,S,D]
      uint16_t* __restrict__ Vtb,        // [B,H,D,S]
      float* __restrict__ present)       // [B,2,H,S,D]
{
  __shared__ __align__(16) uint16_t wtile[2][64 * 32];

  const int tid   = threadIdx.x;
  const int lane  = tid & 31;
  const int wave  = tid >> 5;
  const int m0    = blockIdx.x * 128 + wave * 32;
  const int n0    = blockIdx.y * 64;
  const int which = blockIdx.z;
  const uint16_t* W = (which == 0) ? Wqb : (which == 1) ? Wkb : Wvb;

  stage_w(W, wtile[0], n0, 0, tid);
  v16bf a0 = load_frag16(Xb + (size_t)m0 * 1024, 1024, lane);
  v16bf a1 = load_frag16(Xb + (size_t)(m0 + 16) * 1024, 1024, lane);

  v8f acc[2][4];
#pragma unroll
  for (int s = 0; s < 2; ++s)
#pragma unroll
    for (int j = 0; j < 4; ++j) acc[s][j] = zero8();

#pragma unroll 2
  for (int k = 0, it = 0; k < 1024; k += 32, ++it) {
    const int  cur     = it & 1;
    const bool hasNext = (k + 32) < 1024;
    if (hasNext) stage_w(W, wtile[cur ^ 1], n0, k + 32, tid);
    if (hasNext) wait_async_le2(); else wait_async_0();
    __syncthreads();                               // staged tile visible

    // A prefetch after the barrier: latency hides under the WMMAs below.
    const int kn = hasNext ? k + 32 : k;
    v16bf na0 = load_frag16(Xb + (size_t)m0 * 1024 + kn, 1024, lane);
    v16bf na1 = load_frag16(Xb + (size_t)(m0 + 16) * 1024 + kn, 1024, lane);

    v16bf bfr[4];
#pragma unroll
    for (int j = 0; j < 4; ++j)
      bfr[j] = load_frag16(wtile[cur] + j * 16 * 32, 32, lane);
#pragma unroll
    for (int j = 0; j < 4; ++j) {
      acc[0][j] = wmma_bf16(a0, bfr[j], acc[0][j]);
      acc[1][j] = wmma_bf16(a1, bfr[j], acc[1][j]);
    }
    __syncthreads();                               // reads done before restage
    a0 = na0; a1 = na1;
  }

  const int b    = m0 >> 11;
  const int s0   = m0 & 2047;
  const int h    = blockIdx.y;                     // 64 cols == one head
  const int half = lane >> 4;
  const int nl   = lane & 15;
#pragma unroll
  for (int sub = 0; sub < 2; ++sub) {
#pragma unroll
    for (int j = 0; j < 4; ++j) {
#pragma unroll
      for (int v = 0; v < 8; ++v) {
        const int s = s0 + sub * 16 + v + 8 * half;
        const int d = j * 16 + nl;
        const float val = acc[sub][j][v];
        if (which == 0) {
          Qb[((size_t)(b * 16 + h) * 2048 + s) * 64 + d] = f2bf(val);
        } else if (which == 1) {
          Kb[((size_t)(b * 16 + h) * 2048 + s) * 64 + d] = f2bf(val);
          present[((size_t)((b * 2 + 0) * 16 + h) * 2048 + s) * 64 + d] = val;
        } else {
          Vtb[((size_t)(b * 16 + h) * 64 + d) * 2048 + s] = f2bf(val);
          present[((size_t)((b * 2 + 1) * 16 + h) * 2048 + s) * 64 + d] = val;
        }
      }
    }
  }
}

// ---------------------------------------------------------------------------
// Causal attention. grid (S/64, B*H), block 128; wave owns 16 query rows.
__global__ void __launch_bounds__(128, 1)
k_attn(const uint16_t* __restrict__ Qb,   // [B,H,S,D]
       const uint16_t* __restrict__ Kb,   // [B,H,S,D]
       const uint16_t* __restrict__ Vtb,  // [B,H,D,S]
       float* __restrict__ attn,          // [B*H,S,S]
       uint16_t* __restrict__ resb)       // [B*S,E] bf16
{
  __shared__ __align__(16) uint16_t ldsP[4][16 * 32];

  const int lane = threadIdx.x & 31;
  const int wave = threadIdx.x >> 5;
  const int bh   = blockIdx.y;
  const int b    = bh >> 4;
  const int h    = bh & 15;
  const int qt   = blockIdx.x * 4 + wave;
  const int q0   = qt * 16;
  const int half = lane >> 4;
  const int nl   = lane & 15;

  const uint16_t* Qh = Qb  + (size_t)bh * 2048 * 64;
  const uint16_t* Kh = Kb  + (size_t)bh * 2048 * 64;
  const uint16_t* Vh = Vtb + (size_t)bh * 64 * 2048;

  const v16bf qa0 = load_frag16(Qh + q0 * 64 +  0, 64, lane);
  const v16bf qa1 = load_frag16(Qh + q0 * 64 + 32, 64, lane);

  float pmax[8], psum[8];
#pragma unroll
  for (int v = 0; v < 8; ++v) { pmax[v] = -1e30f; psum[v] = 0.f; }

  // ---- pass 1: per-row max / sum(exp), next K tile prefetched ----
  v16bf kb0 = load_frag16(Kh +  0, 64, lane);
  v16bf kb1 = load_frag16(Kh + 32, 64, lane);
#pragma unroll 2
  for (int kt = 0; kt <= qt; ++kt) {
    const int ktn = (kt < qt) ? kt + 1 : kt;
    v16bf nb0 = load_frag16(Kh + ktn * 1024 +  0, 64, lane);
    v16bf nb1 = load_frag16(Kh + ktn * 1024 + 32, 64, lane);
    v8f c = wmma_bf16(qa0, kb0, zero8());
    c     = wmma_bf16(qa1, kb1, c);
    const int kcol = kt * 16 + nl;
#pragma unroll
    for (int v = 0; v < 8; ++v) {
      const int  q     = q0 + v + 8 * half;
      const bool valid = (kcol <= q);
      const float s    = valid ? c[v] : -1e30f;
      const float nm   = fmaxf(pmax[v], s);
      psum[v] = psum[v] * __expf(pmax[v] - nm) + (valid ? __expf(s - nm) : 0.f);
      pmax[v] = nm;
    }
    kb0 = nb0; kb1 = nb1;
  }
  // reduce across the 16 lanes of each half (masks stay inside the half)
#pragma unroll
  for (int v = 0; v < 8; ++v) {
#pragma unroll
    for (int mask = 1; mask < 16; mask <<= 1) {
      const float om = __shfl_xor(pmax[v], mask);
      const float os = __shfl_xor(psum[v], mask);
      const float nm = fmaxf(pmax[v], om);
      psum[v] = psum[v] * __expf(pmax[v] - nm) + os * __expf(om - nm);
      pmax[v] = nm;
    }
    psum[v] = 0.125f / psum[v];   // fold in post-softmax 1/sqrt(D)
  }

  // ---- pass 2: attn tiles + O = P @ V (loads hoisted ahead of exp work) ---
  v8f o[4] = {zero8(), zero8(), zero8(), zero8()};
  float* attnT = attn + (size_t)bh * 2048 * 2048;
  uint16_t* ldsp = ldsP[wave];
  const int kcMax = (q0 + 15) >> 5;

  for (int kc = 0; kc < 64; ++kc) {
    if (kc <= kcMax) {
      const v16bf kf0a = load_frag16(Kh + (kc * 2 + 0) * 1024 +  0, 64, lane);
      const v16bf kf0b = load_frag16(Kh + (kc * 2 + 0) * 1024 + 32, 64, lane);
      const v16bf kf1a = load_frag16(Kh + (kc * 2 + 1) * 1024 +  0, 64, lane);
      const v16bf kf1b = load_frag16(Kh + (kc * 2 + 1) * 1024 + 32, 64, lane);
      v16bf vf[4];
#pragma unroll
      for (int j = 0; j < 4; ++j)
        vf[j] = load_frag16(Vh + (size_t)(j * 16) * 2048 + kc * 32, 2048, lane);

      v8f c0 = wmma_bf16(qa0, kf0a, zero8());
      c0     = wmma_bf16(qa1, kf0b, c0);
      v8f c1 = wmma_bf16(qa0, kf1a, zero8());
      c1     = wmma_bf16(qa1, kf1b, c1);

#pragma unroll
      for (int sub = 0; sub < 2; ++sub) {
        const v8f& c = sub ? c1 : c0;
        const int kcol = (kc * 2 + sub) * 16 + nl;
#pragma unroll
        for (int v = 0; v < 8; ++v) {
          const int  m     = v + 8 * half;
          const int  q     = q0 + m;
          const bool valid = (kcol <= q);
          const float p = valid ? __expf(c[v] - pmax[v]) * psum[v] : 0.f;
          attnT[(size_t)q * 2048 + kcol] = p;
          ldsp[m * 32 + sub * 16 + nl] = f2bf(p);
        }
      }
      asm volatile("s_wait_dscnt 0" ::: "memory");   // LDS stores visible
      const v16bf pa = load_frag16(ldsp, 32, lane);  // P tile as A-fragment
#pragma unroll
      for (int j = 0; j < 4; ++j) o[j] = wmma_bf16(pa, vf[j], o[j]);
      asm volatile("s_wait_dscnt 0" ::: "memory");   // WAR before overwrite
    } else {
      // strict upper triangle: exact zeros
#pragma unroll
      for (int sub = 0; sub < 2; ++sub) {
        const int kcol = kc * 32 + sub * 16 + nl;
#pragma unroll
        for (int v = 0; v < 8; ++v) {
          const int q = q0 + v + 8 * half;
          attnT[(size_t)q * 2048 + kcol] = 0.f;
        }
      }
    }
  }

#pragma unroll
  for (int j = 0; j < 4; ++j)
#pragma unroll
    for (int v = 0; v < 8; ++v) {
      const int m = v + 8 * half;
      resb[((size_t)(b * 2048 + q0 + m)) * 1024 + h * 64 + j * 16 + nl] =
          f2bf(o[j][v]);
    }
}

// ---------------------------------------------------------------------------
// Output projection: out = res @ Wp.T (f32). grid (32,16), block 128.
// Same LDS-staged, double-buffered structure as k_qkv.
__global__ void __launch_bounds__(128, 1)
k_proj(const uint16_t* __restrict__ Rb,   // [4096,1024] bf16
       const uint16_t* __restrict__ Wpb,  // [1024,1024] bf16
       float* __restrict__ out)           // [4096,1024] f32
{
  __shared__ __align__(16) uint16_t wtile[2][64 * 32];

  const int tid  = threadIdx.x;
  const int lane = tid & 31;
  const int wave = tid >> 5;
  const int m0   = blockIdx.x * 128 + wave * 32;
  const int n0   = blockIdx.y * 64;

  stage_w(Wpb, wtile[0], n0, 0, tid);
  v16bf a0 = load_frag16(Rb + (size_t)m0 * 1024, 1024, lane);
  v16bf a1 = load_frag16(Rb + (size_t)(m0 + 16) * 1024, 1024, lane);

  v8f acc[2][4];
#pragma unroll
  for (int s = 0; s < 2; ++s)
#pragma unroll
    for (int j = 0; j < 4; ++j) acc[s][j] = zero8();

#pragma unroll 2
  for (int k = 0, it = 0; k < 1024; k += 32, ++it) {
    const int  cur     = it & 1;
    const bool hasNext = (k + 32) < 1024;
    if (hasNext) stage_w(Wpb, wtile[cur ^ 1], n0, k + 32, tid);
    if (hasNext) wait_async_le2(); else wait_async_0();
    __syncthreads();

    const int kn = hasNext ? k + 32 : k;
    v16bf na0 = load_frag16(Rb + (size_t)m0 * 1024 + kn, 1024, lane);
    v16bf na1 = load_frag16(Rb + (size_t)(m0 + 16) * 1024 + kn, 1024, lane);

    v16bf bfr[4];
#pragma unroll
    for (int j = 0; j < 4; ++j)
      bfr[j] = load_frag16(wtile[cur] + j * 16 * 32, 32, lane);
#pragma unroll
    for (int j = 0; j < 4; ++j) {
      acc[0][j] = wmma_bf16(a0, bfr[j], acc[0][j]);
      acc[1][j] = wmma_bf16(a1, bfr[j], acc[1][j]);
    }
    __syncthreads();
    a0 = na0; a1 = na1;
  }

  const int half = lane >> 4;
  const int nl   = lane & 15;
#pragma unroll
  for (int sub = 0; sub < 2; ++sub)
#pragma unroll
    for (int j = 0; j < 4; ++j)
#pragma unroll
      for (int v = 0; v < 8; ++v)
        out[(size_t)(m0 + sub * 16 + v + 8 * half) * 1024 + n0 + j * 16 + nl] =
            acc[sub][j][v];
}

// ---------------------------------------------------------------------------
extern "C" void kernel_launch(void* const* d_in, const int* in_sizes, int n_in,
                              void* d_out, int out_size, void* d_ws,
                              size_t ws_size, hipStream_t stream) {
  (void)in_sizes; (void)n_in; (void)out_size; (void)ws_size;
  const float* X  = (const float*)d_in[0];
  const float* Wq = (const float*)d_in[1];
  const float* Wk = (const float*)d_in[2];
  const float* Wv = (const float*)d_in[3];
  const float* Wp = (const float*)d_in[4];
  float* out = (float*)d_out;

  float* outO  = out;                          //  4,194,304 f32
  float* attnO = out + 4194304;                // 134,217,728 f32
  float* presO = out + 4194304 + 134217728;    //  8,388,608 f32

  char* ws = (char*)d_ws;                      // ~48 MB staging
  uint16_t* Xb   = (uint16_t*)(ws +  0ull * (1ull << 20));
  uint16_t* Wqb  = (uint16_t*)(ws +  8ull * (1ull << 20));
  uint16_t* Wkb  = (uint16_t*)(ws + 10ull * (1ull << 20));
  uint16_t* Wvb  = (uint16_t*)(ws + 12ull * (1ull << 20));
  uint16_t* Wpb  = (uint16_t*)(ws + 14ull * (1ull << 20));
  uint16_t* Qb   = (uint16_t*)(ws + 16ull * (1ull << 20));
  uint16_t* Kb   = (uint16_t*)(ws + 24ull * (1ull << 20));
  uint16_t* Vtb  = (uint16_t*)(ws + 32ull * (1ull << 20));
  uint16_t* resb = (uint16_t*)(ws + 40ull * (1ull << 20));

  k_f32_to_bf16<<<(4194304 + 255) / 256, 256, 0, stream>>>(X,  Xb,  4194304);
  k_f32_to_bf16<<<(1048576 + 255) / 256, 256, 0, stream>>>(Wq, Wqb, 1048576);
  k_f32_to_bf16<<<(1048576 + 255) / 256, 256, 0, stream>>>(Wk, Wkb, 1048576);
  k_f32_to_bf16<<<(1048576 + 255) / 256, 256, 0, stream>>>(Wv, Wvb, 1048576);
  k_f32_to_bf16<<<(1048576 + 255) / 256, 256, 0, stream>>>(Wp, Wpb, 1048576);

  k_qkv <<<dim3(32, 16, 3), 128, 0, stream>>>(Xb, Wqb, Wkb, Wvb,
                                              Qb, Kb, Vtb, presO);
  k_attn<<<dim3(32, 32),    128, 0, stream>>>(Qb, Kb, Vtb, attnO, resb);
  k_proj<<<dim3(32, 16),    128, 0, stream>>>(resb, Wpb, outO);
}
